// LCGLoss_51316269253343
// MI455X (gfx1250) — matrix operand order repaired and checked by reading, not compiled, not run on hardware
//
#include <hip/hip_runtime.h>
#include <hip/hip_bf16.h>

#define BB 128
#define LL 1024
#define DD 512
#define MT 64     // rows per workgroup
#define EC 64     // E-chunk width
#define NCH (DD/EC)
#define SSTR 132  // padded score-tile row stride (floats) to avoid bank conflicts

typedef __attribute__((ext_vector_type(16))) __bf16 v16bf;
typedef __attribute__((ext_vector_type(8)))  __bf16 v8bf;
typedef __attribute__((ext_vector_type(8)))  float  v8f;

union V16U { v16bf v; v8bf h[2]; };

__device__ __forceinline__ __bf16 f2bf(float x) {
    unsigned u = __builtin_bit_cast(unsigned, x);
    unsigned r = (u + 0x7fffu + ((u >> 16) & 1u)) >> 16;
    return __builtin_bit_cast(__bf16, (unsigned short)r);
}

// A-fragment (16x32 bf16): lanes 0-15 hold row M=ln with K k0..k0+15,
// lanes 16-31 hold same rows with K k0+16..k0+31.
__device__ __forceinline__ v16bf load_frag_A(const __bf16* row, int k0, int half) {
    V16U u;
    u.h[0] = *(const v8bf*)(row + k0 + half * 16);
    u.h[1] = *(const v8bf*)(row + k0 + half * 16 + 8);
    return u.v;
}

// B-fragment (32x16 bf16): lanes 0-15 hold col N=ln with K k0..k0+15,
// lanes 16-31 hold K k0+16..k0+31. Source row is K-contiguous.
__device__ __forceinline__ v16bf load_frag_B(const __bf16* row, int k0, int half) {
    const v8bf* p = (const v8bf*)(row + k0 + half * 16);
    V16U u;
    u.h[0] = p[0];
    u.h[1] = p[1];
    return u.v;
}

__device__ __forceinline__ v8f wmma_bf16(v16bf a, v16bf b, v8f c) {
    return __builtin_amdgcn_wmma_f32_16x16x32_bf16(false, a, false, b, (short)0, c, false, false);
}

__global__ __launch_bounds__(256) void k_init(float* g_cross, float* g_accum) {
    int t = threadIdx.x;
    if (t < BB) g_cross[t] = 0.0f;
    if (t < 2)  g_accum[t] = 0.0f;
}

// One-shot: W_ML fp32 -> bf16 (L2-resident copy used by every workgroup of k_main)
__global__ __launch_bounds__(256) void k_wconv(const float* __restrict__ W,
                                               __bf16* __restrict__ out) {
    size_t i = ((size_t)blockIdx.x * 256 + threadIdx.x) * 4;
    float4 v = *(const float4*)(W + i);
    union { __bf16 h[4]; uint2 u; } p;
    p.h[0] = f2bf(v.x); p.h[1] = f2bf(v.y); p.h[2] = f2bf(v.z); p.h[3] = f2bf(v.w);
    *(uint2*)(out + i) = p.u;
}

// image_proj[b,e] = sum_d image[b,d] * W_MV[e,d]  (f32 accumulate, store bf16)
__global__ __launch_bounds__(256) void k_img(const float* __restrict__ img,
                                             const float* __restrict__ W_MV,
                                             __bf16* __restrict__ imgp) {
    __shared__ float srow[DD];
    int b = blockIdx.x;
    for (int i = threadIdx.x; i < DD; i += 256) srow[i] = img[(size_t)b * DD + i];
    __syncthreads();
    for (int e = threadIdx.x; e < DD; e += 256) {
        const float4* w = (const float4*)(W_MV + (size_t)e * DD);
        float s = 0.0f;
        for (int d = 0; d < DD / 4; ++d) {
            float4 v = w[d];
            const float* r = &srow[d * 4];
            s += v.x * r[0] + v.y * r[1] + v.z * r[2] + v.w * r[3];
        }
        imgp[(size_t)b * DD + e] = f2bf(s);
    }
}

// Fused: proj-tile GEMM (bf16 WMMA) + score GEMM + exp/reductions.
__global__ __launch_bounds__(256) void k_main(const float* __restrict__ text,
                                              const unsigned char* __restrict__ mask,
                                              const float* __restrict__ taup,
                                              const __bf16* __restrict__ g_wbf,
                                              const __bf16* __restrict__ imgp,
                                              float* __restrict__ g_epos,
                                              float* __restrict__ g_denom,
                                              float* __restrict__ g_cross) {
    __shared__ __align__(16) __bf16 lds_text[MT][DD];            // 64 KB
    __shared__ __align__(16) unsigned char smem_wS[EC * DD * 2]; // 64 KB (W chunk | score tile)
    __shared__ __align__(16) __bf16 lds_p[MT][EC];               // 8 KB
    __shared__ unsigned char s_rowvalid[MT];

    __bf16 (*lds_w)[DD]  = (__bf16(*)[DD])smem_wS;
    float  (*lds_S)[SSTR] = (float(*)[SSTR])smem_wS;

    const int tid  = threadIdx.x;
    const int wave = tid >> 5;
    const int lane = tid & 31;
    const int half = lane >> 4;
    const int ln   = lane & 15;

    const int m0     = blockIdx.x * MT;
    const int b_self = m0 >> 10;     // /L
    const int l0     = m0 & (LL - 1);

    // Stage text tile as bf16 in LDS (fp32 read from HBM happens exactly once)
    const float* tbase = text + (size_t)m0 * DD;
    for (int i = tid * 4; i < MT * DD; i += 256 * 4) {
        float4 v = *(const float4*)(tbase + i);
        __bf16* d = ((__bf16*)lds_text) + i;
        d[0] = f2bf(v.x); d[1] = f2bf(v.y); d[2] = f2bf(v.z); d[3] = f2bf(v.w);
    }

    const int mt = wave & 3;   // 16-row group for both phases
    const int eh = wave >> 2;  // phase A: 32-col E half; phase B: 64-col image half

    v8f zero;
    for (int i = 0; i < 8; ++i) zero[i] = 0.0f;
    v8f Sacc[4];
    for (int t = 0; t < 4; ++t) Sacc[t] = zero;

    for (int ch = 0; ch < NCH; ++ch) {
        const int e0 = ch * EC;
        __syncthreads();
        // Stage pre-converted bf16 W chunk: pure b128 copy, no conversion
        {
            const uint4* wsrc = (const uint4*)(g_wbf + (size_t)e0 * DD);
            uint4* wdst = (uint4*)smem_wS;
            for (int i = tid; i < EC * DD / 8; i += 256)
                wdst[i] = wsrc[i];
        }
        __syncthreads();

        // Warm L2/WGP$ for the NEXT chunk's W while this chunk computes
        if (ch + 1 < NCH) {
            const char* nb = (const char*)(g_wbf + (size_t)(e0 + EC) * DD);
            __builtin_prefetch(nb + tid * 256, 0, 0);
            __builtin_prefetch(nb + tid * 256 + 128, 0, 0);
        }

        // Issue phase-B B-fragment global loads now; they only depend on e0 and
        // complete under phase-A's WMMA work.
        v16bf bfr[2][4];
        for (int kk = 0; kk < 2; ++kk)
            for (int it = 0; it < 4; ++it) {
                int g = eh * 64 + it * 16 + ln;
                bfr[kk][it] = load_frag_B(imgp + (size_t)g * DD + e0, kk * 32, half);
            }

        // Phase A: proj chunk [64m x 64e]; wave tile = 16m x 32e (2 WMMA tiles),
        // software-pipelined fragment loads.
        v8f acc0 = zero, acc1 = zero;
        const __bf16* arow  = &lds_text[16 * mt + ln][0];
        const __bf16* brow0 = &lds_w[32 * eh + ln][0];
        const __bf16* brow1 = &lds_w[32 * eh + 16 + ln][0];
        v16bf a  = load_frag_A(arow,  0, half);
        v16bf b0 = load_frag_B(brow0, 0, half);
        v16bf b1 = load_frag_B(brow1, 0, half);
        for (int k0 = 0; k0 < DD; k0 += 32) {
            v16bf an = a, b0n = b0, b1n = b1;
            if (k0 + 32 < DD) {
                an  = load_frag_A(arow,  k0 + 32, half);
                b0n = load_frag_B(brow0, k0 + 32, half);
                b1n = load_frag_B(brow1, k0 + 32, half);
            }
            acc0 = wmma_bf16(a, b0, acc0);
            acc1 = wmma_bf16(a, b1, acc1);
            a = an; b0 = b0n; b1 = b1n;
        }
        // Spill proj chunk to LDS as bf16 (C/D layout: VGPR r -> M=r(+8*half), N=ln)
        for (int r = 0; r < 8; ++r) {
            int m = 16 * mt + r + 8 * half;
            lds_p[m][32 * eh + ln]      = f2bf(acc0[r]);
            lds_p[m][32 * eh + 16 + ln] = f2bf(acc1[r]);
        }
        __syncthreads();

        // Phase B: S[64m x 128img] += proj_chunk @ image_proj^T ; wave tile = 16m x 64img
        const __bf16* prow = &lds_p[16 * mt + ln][0];
        for (int kk = 0; kk < 2; ++kk) {
            v16bf pa = load_frag_A(prow, kk * 32, half);
            for (int it = 0; it < 4; ++it)
                Sacc[it] = wmma_bf16(pa, bfr[kk][it], Sacc[it]);
        }
    }
    __syncthreads();

    // Dump score tile to (bank-conflict-padded) LDS; aliases the W buffer.
    for (int it = 0; it < 4; ++it)
        for (int r = 0; r < 8; ++r)
            lds_S[16 * mt + r + 8 * half][eh * 64 + it * 16 + ln] = Sacc[it][r];
    if (tid < MT)
        s_rowvalid[tid] = (mask[(size_t)b_self * LL + l0 + tid] == 0) ? 1 : 0;
    __syncthreads();

    float tau  = taup[0];
    tau        = fminf(fmaxf(tau, 0.05f), 2.0f);
    float invt = 1.0f / tau;

    // Row pass: exp, denom (masked by valid[k,l]), e_pos (k == b_self)
    if (tid < MT) {
        int m = tid, l = l0 + m;
        float denom = 0.0f, epos = 0.0f;
        for (int k = 0; k < BB; ++k) {
            float e = __expf(lds_S[m][k] * invt);
            lds_S[m][k] = e;
            bool vkl = (mask[(size_t)k * LL + l] == 0);
            denom += vkl ? e : 0.0f;
            if (k == b_self) epos = e;
        }
        g_denom[(size_t)b_self * LL + l] = denom;
        g_epos [(size_t)b_self * LL + l] = epos;
    }
    __syncthreads();

    // Column pass: cross_sum[i] += sum over valid own rows, i != b_self
    if (tid < BB && tid != b_self) {
        float c = 0.0f;
        for (int m = 0; m < MT; ++m)
            c += s_rowvalid[m] ? lds_S[m][tid] : 0.0f;
        atomicAdd(&g_cross[tid], c);
    }
}

__global__ __launch_bounds__(256) void k_loss(const float* __restrict__ g_epos,
                                              const float* __restrict__ g_denom,
                                              const float* __restrict__ g_cross,
                                              const unsigned char* __restrict__ mask,
                                              float* __restrict__ g_accum) {
    __shared__ float s_num[256];
    __shared__ float s_cnt[256];
    float ln = 0.0f, lc = 0.0f;
    for (size_t i = (size_t)blockIdx.x * 256 + threadIdx.x; i < (size_t)BB * LL;
         i += (size_t)gridDim.x * 256) {
        if (mask[i] == 0) {
            int b = (int)(i >> 10);
            float ep = g_epos[i];
            float f1 = ep / g_denom[i];
            float f2 = ep / (ep + g_cross[b]);
            ln += 0.5f * (f1 + f2);
            lc += 1.0f;
        }
    }
    s_num[threadIdx.x] = ln;
    s_cnt[threadIdx.x] = lc;
    __syncthreads();
    for (int s = 128; s > 0; s >>= 1) {
        if ((int)threadIdx.x < s) {
            s_num[threadIdx.x] += s_num[threadIdx.x + s];
            s_cnt[threadIdx.x] += s_cnt[threadIdx.x + s];
        }
        __syncthreads();
    }
    if (threadIdx.x == 0) {
        atomicAdd(&g_accum[0], s_num[0]);
        atomicAdd(&g_accum[1], s_cnt[0]);
    }
}

__global__ void k_final(const float* __restrict__ g_accum, float* __restrict__ out) {
    out[0] = g_accum[0] / g_accum[1];
}

extern "C" void kernel_launch(void* const* d_in, const int* in_sizes, int n_in,
                              void* d_out, int out_size, void* d_ws, size_t ws_size,
                              hipStream_t stream) {
    (void)in_sizes; (void)n_in; (void)out_size; (void)ws_size;
    const float*         text  = (const float*)d_in[0];          // [B,L,D]
    const float*         img   = (const float*)d_in[1];          // [B,D]
    const unsigned char* mask  = (const unsigned char*)d_in[2];  // [B,L] bool
    const float*         tau   = (const float*)d_in[3];          // [1]
    const float*         W_ML  = (const float*)d_in[4];          // [D,D]
    const float*         W_MV  = (const float*)d_in[5];          // [D,D]

    char* ws = (char*)d_ws;
    float*  g_cross = (float*)ws;                                   // 128 f32
    float*  g_accum = (float*)(ws + 512);                           // 2 f32
    __bf16* g_imgp  = (__bf16*)(ws + 1024);                         // 128*512 bf16
    float*  g_epos  = (float*)(ws + 1024 + BB * DD * 2);            // 128*1024 f32
    float*  g_denom = (float*)(ws + 1024 + BB * DD * 2 + (size_t)BB * LL * 4);
    __bf16* g_wbf   = (__bf16*)(ws + 1024 + BB * DD * 2 + (size_t)BB * LL * 8); // 512*512 bf16

    k_init<<<1, 256, 0, stream>>>(g_cross, g_accum);
    k_wconv<<<(DD * DD) / (4 * 256), 256, 0, stream>>>(W_ML, g_wbf);
    k_img<<<BB, 256, 0, stream>>>(img, W_MV, g_imgp);
    k_main<<<(BB * LL) / MT, 256, 0, stream>>>(text, mask, tau, g_wbf, g_imgp,
                                               g_epos, g_denom, g_cross);
    k_loss<<<256, 256, 0, stream>>>(g_epos, g_denom, g_cross, mask, g_accum);
    k_final<<<1, 1, 0, stream>>>(g_accum, (float*)d_out);
}
